// HBitLinear_850403524834
// MI455X (gfx1250) — 8-bit matrix operand order repaired and checked by reading, compiled, not run on hardware
//
#include <hip/hip_runtime.h>
#include <stdint.h>

typedef int v8i __attribute__((ext_vector_type(8)));
typedef int v4int __attribute__((vector_size(16)));

#define D      4096
#define NROWS  8192
#define NOUT   4096
#define BM     128
#define BN     128
#define BK     64
#define LDT    80
#define TILEB  (BM * LDT)

#if defined(__AMDGCN__) && defined(__has_builtin)
#if __has_builtin(__builtin_amdgcn_global_load_async_to_lds_b128)
#define USE_ASYNC_LDS 1
#endif
#endif
#ifndef USE_ASYNC_LDS
#define USE_ASYNC_LDS 0
#endif

__device__ __forceinline__ void wait_async0() {
#if defined(__has_builtin)
#if __has_builtin(__builtin_amdgcn_s_wait_asynccnt)
    __builtin_amdgcn_s_wait_asynccnt(0);
#else
    asm volatile("s_wait_asynccnt 0x0" ::: "memory");
#endif
#endif
}

#if USE_ASYNC_LDS
__device__ __forceinline__ void cp16_async(signed char* lds, const signed char* g) {
    __attribute__((address_space(3))) v4int* l =
        (__attribute__((address_space(3))) v4int*)(uint32_t)(uintptr_t)lds;
    __attribute__((address_space(1))) v4int* gp =
        (__attribute__((address_space(1))) v4int*)(uintptr_t)g;
    __builtin_amdgcn_global_load_async_to_lds_b128(gp, l, 0, 0);
}
#endif

__global__ void k_wabs_partial(const float* __restrict__ w,
                               float* __restrict__ partials, int n) {
    __shared__ float red[8];
    float s = 0.f;
    for (int i = blockIdx.x * blockDim.x + threadIdx.x; i < n; i += gridDim.x * blockDim.x)
        s += fabsf(w[i]);
    for (int off = 16; off > 0; off >>= 1) s += __shfl_down(s, off);
    int wid = threadIdx.x >> 5, lane = threadIdx.x & 31;
    if (lane == 0) red[wid] = s;
    __syncthreads();
    if (threadIdx.x == 0) {
        float t = 0.f;
        for (int i = 0; i < 8; ++i) t += red[i];
        partials[blockIdx.x] = t;
    }
}

__global__ void k_wscale_final(const float* __restrict__ partials, int np,
                               float* __restrict__ wscale, float inv_n) {
    __shared__ float red[8];
    float s = 0.f;
    for (int i = threadIdx.x; i < np; i += blockDim.x) s += partials[i];
    for (int off = 16; off > 0; off >>= 1) s += __shfl_down(s, off);
    int wid = threadIdx.x >> 5, lane = threadIdx.x & 31;
    if (lane == 0) red[wid] = s;
    __syncthreads();
    if (threadIdx.x == 0) {
        float t = 0.f;
        for (int i = 0; i < 8; ++i) t += red[i];
        wscale[0] = t * inv_n;
    }
}

__global__ void k_wquant(const float* __restrict__ w,
                         const float* __restrict__ wscale,
                         signed char* __restrict__ wq, int n) {
    float half_s = 0.5f * wscale[0];
    for (int i = blockIdx.x * blockDim.x + threadIdx.x; i < n; i += gridDim.x * blockDim.x) {
        float v = w[i];
        wq[i] = (v > half_s) ? (signed char)1 : ((v < -half_s) ? (signed char)-1 : (signed char)0);
    }
}

__global__ __launch_bounds__(256) void k_act(const float* __restrict__ x,
                                             const float* __restrict__ gamma,
                                             const float* __restrict__ beta,
                                             signed char* __restrict__ alo,
                                             signed char* __restrict__ ahi) {
    __shared__ float buf[D];
    __shared__ float red[8], red2[8];
    __shared__ float sc[2];
    const int t = threadIdx.x;
    const int wid = t >> 5, lane = t & 31;
    const size_t base = (size_t)blockIdx.x * D;

    float sum = 0.f, sumsq = 0.f;
#pragma unroll
    for (int j = 0; j < 16; ++j) {
        int idx = t + 256 * j;
        float v = x[base + idx];
        buf[idx] = v;
        sum += v; sumsq += v * v;
    }
    for (int off = 16; off > 0; off >>= 1) {
        sum   += __shfl_down(sum, off);
        sumsq += __shfl_down(sumsq, off);
    }
    if (lane == 0) { red[wid] = sum; red2[wid] = sumsq; }
    __syncthreads();
    if (t == 0) {
        float S = 0.f, S2 = 0.f;
        for (int i = 0; i < 8; ++i) { S += red[i]; S2 += red2[i]; }
        float mean = S * (1.0f / D);
        float var  = S2 * (1.0f / D) - mean * mean;
        sc[0] = mean;
        sc[1] = rsqrtf(var + 1e-5f);
    }
    __syncthreads();
    float mean = sc[0], rstd = sc[1];

    float mx = 0.f;
#pragma unroll
    for (int j = 0; j < 16; ++j) {
        int idx = t + 256 * j;
        float v = (buf[idx] - mean) * rstd * gamma[idx] + beta[idx];
        buf[idx] = v;
        mx = fmaxf(mx, fabsf(v));
    }
    for (int off = 16; off > 0; off >>= 1) mx = fmaxf(mx, __shfl_down(mx, off));
    if (lane == 0) red[wid] = mx;
    __syncthreads();
    if (t == 0) {
        float m = 0.f;
        for (int i = 0; i < 8; ++i) m = fmaxf(m, red[i]);
        float scale = fminf(fmaxf(m, 1e-6f), 1e6f);
        sc[0] = fminf(fmaxf(7.0f / scale, 1e-6f), 1e6f);
    }
    __syncthreads();
    float sf = sc[0];

#pragma unroll
    for (int j = 0; j < 16; ++j) {
        int idx = t + 256 * j;
        buf[idx] = fminf(fmaxf(roundf(buf[idx] * sf), -7.f), 7.f);
    }
    __syncthreads();

    for (int h = 1; h < D; h <<= 1) {
#pragma unroll
        for (int j = 0; j < 8; ++j) {
            int p = t + 256 * j;
            int i = ((p / h) * (h << 1)) + (p % h);
            float a = buf[i], b = buf[i + h];
            buf[i] = a + b;
            buf[i + h] = a - b;
        }
        __syncthreads();
    }

#pragma unroll
    for (int j = 0; j < 16; ++j) {
        int idx = t + 256 * j;
        int xi = (int)buf[idx];
        int hi = (xi + 128) >> 8;
        int lo = xi - (hi << 8);
        alo[base + idx] = (signed char)lo;
        ahi[base + idx] = (signed char)hi;
    }
}

__device__ __forceinline__ v8i load_a_frag(const signed char* p) {
    int2 u0 = *(const int2*)(p);
    int2 u1 = *(const int2*)(p + 16);
    int2 u2 = *(const int2*)(p + 32);
    int2 u3 = *(const int2*)(p + 48);
    v8i a;
    a[0] = u0.x; a[1] = u0.y; a[2] = u1.x; a[3] = u1.y;
    a[4] = u2.x; a[5] = u2.y; a[6] = u3.x; a[7] = u3.y;
    return a;
}

__device__ __forceinline__ v8i load_b_frag(const signed char* p) {
    int2 q0 = *(const int2*)(p);
    int2 q1 = *(const int2*)(p + 16);
    int2 q2 = *(const int2*)(p + 32);
    int2 q3 = *(const int2*)(p + 48);
    v8i b;
    b[0] = q0.x; b[1] = q0.y; b[2] = q1.x; b[3] = q1.y;
    b[4] = q2.x; b[5] = q2.y; b[6] = q3.x; b[7] = q3.y;
    return b;
}

__global__ __launch_bounds__(256) void k_gemm(const signed char* __restrict__ alo,
                                              const signed char* __restrict__ ahi,
                                              const signed char* __restrict__ wq,
                                              const float* __restrict__ wscale,
                                              float* __restrict__ out) {
    __shared__ __align__(16) signed char smem[2 * 3 * TILEB];

    const int row0 = blockIdx.y * BM;
    const int col0 = blockIdx.x * BN;
    const int t    = threadIdx.x;
    const int w    = t >> 5;
    const int lane = t & 31;
    const int wm   = w & 3;
    const int wn   = w >> 2;
    const int half = lane >> 4;
    const int l16  = lane & 15;

    const int rr0  = t >> 2;
    const int seg  = (t & 3) * 16;
    const int l0   = rr0 * LDT + seg;
    const int l1   = l0 + 64 * LDT;

    const signed char* ga = alo + (size_t)row0 * D;
    const signed char* gh = ahi + (size_t)row0 * D;
    const signed char* gb = wq  + (size_t)col0 * D;

    signed char* buf0 = smem;
    signed char* buf1 = smem + 3 * TILEB;

    v8i acc_lo[2][4], acc_hi[2][4];
#pragma unroll
    for (int ms = 0; ms < 2; ++ms)
#pragma unroll
        for (int ns = 0; ns < 4; ++ns) { acc_lo[ms][ns] = (v8i)0; acc_hi[ms][ns] = (v8i)0; }

    {
        size_t o0 = (size_t)rr0 * D + seg;
        size_t o1 = o0 + (size_t)64 * D;
#if USE_ASYNC_LDS
        cp16_async(buf0 + l0,             ga + o0);
        cp16_async(buf0 + l1,             ga + o1);
        cp16_async(buf0 + TILEB + l0,     gh + o0);
        cp16_async(buf0 + TILEB + l1,     gh + o1);
        cp16_async(buf0 + 2 * TILEB + l0, gb + o0);
        cp16_async(buf0 + 2 * TILEB + l1, gb + o1);
        wait_async0();
#else
        *(int4*)(buf0 + l0)             = *(const int4*)(ga + o0);
        *(int4*)(buf0 + l1)             = *(const int4*)(ga + o1);
        *(int4*)(buf0 + TILEB + l0)     = *(const int4*)(gh + o0);
        *(int4*)(buf0 + TILEB + l1)     = *(const int4*)(gh + o1);
        *(int4*)(buf0 + 2 * TILEB + l0) = *(const int4*)(gb + o0);
        *(int4*)(buf0 + 2 * TILEB + l1) = *(const int4*)(gb + o1);
#endif
    }

    signed char* cur = buf0;
    signed char* nxt = buf1;
#if !USE_ASYNC_LDS
    int4 rg[6];
#endif

    for (int k0 = 0; k0 < D; k0 += BK) {
        __syncthreads();
        const bool more = (k0 + BK) < D;
        if (more) {
            size_t o0 = (size_t)rr0 * D + (k0 + BK) + seg;
            size_t o1 = o0 + (size_t)64 * D;
#if USE_ASYNC_LDS
            cp16_async(nxt + l0,             ga + o0);
            cp16_async(nxt + l1,             ga + o1);
            cp16_async(nxt + TILEB + l0,     gh + o0);
            cp16_async(nxt + TILEB + l1,     gh + o1);
            cp16_async(nxt + 2 * TILEB + l0, gb + o0);
            cp16_async(nxt + 2 * TILEB + l1, gb + o1);
#else
            rg[0] = *(const int4*)(ga + o0);
            rg[1] = *(const int4*)(ga + o1);
            rg[2] = *(const int4*)(gh + o0);
            rg[3] = *(const int4*)(gh + o1);
            rg[4] = *(const int4*)(gb + o0);
            rg[5] = *(const int4*)(gb + o1);
#endif
        }

        const signed char* slo = cur;
        const signed char* shi = cur + TILEB;
        const signed char* sb  = cur + 2 * TILEB;

        v8i aflo[2], afhi[2];
#pragma unroll
        for (int ms = 0; ms < 2; ++ms) {
            const signed char* pl = slo + (wm * 32 + ms * 16 + l16) * LDT + half * 8;
            const signed char* ph = shi + (wm * 32 + ms * 16 + l16) * LDT + half * 8;
            aflo[ms] = load_a_frag(pl);
            afhi[ms] = load_a_frag(ph);
        }
#pragma unroll
        for (int ns = 0; ns < 4; ++ns) {
            const signed char* pb = sb + (wn * 64 + ns * 16 + l16) * LDT + half * 8;
            v8i bf = load_b_frag(pb);
#pragma unroll
            for (int ms = 0; ms < 2; ++ms) {
                acc_lo[ms][ns] = __builtin_amdgcn_wmma_i32_16x16x64_iu8(
                    true, aflo[ms], true, bf, acc_lo[ms][ns], false, false);
                acc_hi[ms][ns] = __builtin_amdgcn_wmma_i32_16x16x64_iu8(
                    true, afhi[ms], true, bf, acc_hi[ms][ns], false, false);
            }
        }

        if (more) {
#if USE_ASYNC_LDS
            wait_async0();
#else
            *(int4*)(nxt + l0)             = rg[0];
            *(int4*)(nxt + l1)             = rg[1];
            *(int4*)(nxt + TILEB + l0)     = rg[2];
            *(int4*)(nxt + TILEB + l1)     = rg[3];
            *(int4*)(nxt + 2 * TILEB + l0) = rg[4];
            *(int4*)(nxt + 2 * TILEB + l1) = rg[5];
#endif
        }
        signed char* tmp = cur; cur = nxt; nxt = tmp;
    }

    const float s = wscale[0];
#pragma unroll
    for (int ms = 0; ms < 2; ++ms)
#pragma unroll
        for (int ns = 0; ns < 4; ++ns)
#pragma unroll
            for (int r = 0; r < 8; ++r) {
                int v = acc_lo[ms][ns][r] + (acc_hi[ms][ns][r] << 8);
                int grow = row0 + wm * 32 + ms * 16 + r + half * 8;
                int gcol = col0 + wn * 64 + ns * 16 + l16;
                out[(size_t)grow * NOUT + gcol] = (float)v * s;
            }
}

__global__ __launch_bounds__(256) void k_fwht_rows(float* __restrict__ data) {
    __shared__ float buf[D];
    const int t = threadIdx.x;
    const size_t base = (size_t)blockIdx.x * D;
#pragma unroll
    for (int j = 0; j < 16; ++j) {
        int idx = t + 256 * j;
        buf[idx] = data[base + idx];
    }
    __syncthreads();
    for (int h = 1; h < D; h <<= 1) {
#pragma unroll
        for (int j = 0; j < 8; ++j) {
            int p = t + 256 * j;
            int i = ((p / h) * (h << 1)) + (p % h);
            float a = buf[i], b = buf[i + h];
            buf[i] = a + b;
            buf[i + h] = a - b;
        }
        __syncthreads();
    }
#pragma unroll
    for (int j = 0; j < 16; ++j) {
        int idx = t + 256 * j;
        data[base + idx] = buf[idx];
    }
}

extern "C" void kernel_launch(void* const* d_in, const int* in_sizes, int n_in,
                              void* d_out, int out_size, void* d_ws, size_t ws_size,
                              hipStream_t stream) {
    const float* x      = (const float*)d_in[0];
    const float* gamma  = (const float*)d_in[1];
    const float* beta   = (const float*)d_in[2];
    const float* weight = (const float*)d_in[3];
    float* out = (float*)d_out;

    char* ws = (char*)d_ws;
    float*       partials = (float*)ws;
    float*       wscale   = (float*)(ws + 4096);
    signed char* wq       = (signed char*)(ws + 8192);
    signed char* alo      = wq  + (size_t)NOUT * D;
    signed char* ahi      = alo + (size_t)NROWS * D;

    const int nW = NOUT * D;

    k_wabs_partial<<<1024, 256, 0, stream>>>(weight, partials, nW);
    k_wscale_final<<<1, 256, 0, stream>>>(partials, 1024, wscale, 1.0f / (float)nW);
    k_wquant<<<4096, 256, 0, stream>>>(weight, wscale, wq, nW);
    k_act<<<NROWS, 256, 0, stream>>>(x, gamma, beta, alo, ahi);
    dim3 g(NOUT / BN, NROWS / BM);
    k_gemm<<<g, 256, 0, stream>>>(alo, ahi, wq, wscale, out);
    k_fwht_rows<<<NROWS, 256, 0, stream>>>(out);
}
